// RotationInvariantMPFMLP_781684048071
// MI455X (gfx1250) — matrix operand order repaired and checked by reading, not compiled, not run on hardware
//
#include <hip/hip_runtime.h>
#include <hip/hip_bf16.h>

typedef __attribute__((ext_vector_type(16))) __bf16 v16bf;
typedef __attribute__((ext_vector_type(8)))  float  v8f;

union Frag {
  v16bf bf;
  uint4 u[2];
  unsigned d[8];
};

__device__ __forceinline__ unsigned short f2bf(float f) {
  unsigned u = __float_as_uint(f);
  u += 0x7FFFu + ((u >> 16) & 1u);       // round-to-nearest-even
  return (unsigned short)(u >> 16);
}

// ---- flat triu-gather index table (matches np.triu_indices row-major order) ----
struct FlatTbl { int v[64]; };
static constexpr FlatTbl make_flat() {
  FlatTbl F{}; int n = 0;
  for (int r = 0; r < 16; ++r)
    for (int c = r; c < 16; ++c) {
      int d1 = (c - r + 16) & 15, d2 = (r - c + 16) & 15;
      bool m = (r == c) || (d1 >= 1 && d1 <= 3) || (d2 >= 1 && d2 <= 3);
      if (m) F.v[n++] = r * 16 + c;
    }
  return F;
}
__device__ __constant__ FlatTbl FLAT = make_flat();

// =====================================================================
// Kernel 1: repack W1 (512x2048) and W2 (512x512) fp32 -> bf16 B-fragments.
// B-frag layout (32x16 KxN, wave32): lane = N (lanes 16-31 K += 8),
// dwords 0..3: K=0..7, dwords 4..7: K=16..23 (pairs). Frag stored as 32 B per lane, 1024 B per frag.
// =====================================================================
__global__ void __launch_bounds__(256) prep_w_kernel(
    const float* __restrict__ W1, const float* __restrict__ W2,
    unsigned char* __restrict__ WB1, unsigned char* __restrict__ WB2) {
  int gid = blockIdx.x * 256 + threadIdx.x;
  const float* src;
  unsigned char* dst;
  if (gid < 32 * 64 * 32) {                       // W1: 32 ntiles x 64 kchunks
    int lane = gid & 31;
    int t = gid >> 5;                             // t = nt*64 + kc
    int n = ((t >> 6) << 4) + (lane & 15);
    int kbase = ((t & 63) << 5) + ((lane & 16) ? 8 : 0);
    src = W1 + (size_t)n * 2048 + kbase;
    dst = WB1 + ((size_t)t << 10) + lane * 32;
  } else {                                        // W2: 32 ntiles x 16 kchunks
    int g = gid - 32 * 64 * 32;
    if (g >= 32 * 16 * 32) return;
    int lane = g & 31;
    int t = g >> 5;                               // t = nt*16 + kc
    int n = ((t >> 4) << 4) + (lane & 15);
    int kbase = ((t & 15) << 5) + ((lane & 16) ? 8 : 0);
    src = W2 + (size_t)n * 512 + kbase;
    dst = WB2 + ((size_t)t << 10) + lane * 32;
  }
  unsigned d[8];
#pragma unroll
  for (int j = 0; j < 8; ++j) {
    unsigned lo = f2bf(src[(j < 4 ? 2 * j : 8 + 2 * j)]);
    unsigned hi = f2bf(src[(j < 4 ? 2 * j : 8 + 2 * j) + 1]);
    d[j] = lo | (hi << 16);
  }
  uint4* o = (uint4*)dst;
  o[0] = make_uint4(d[0], d[1], d[2], d[3]);
  o[1] = make_uint4(d[4], d[5], d[6], d[7]);
}

// =====================================================================
// Kernel 2: fused roll + gather + fp32->bf16 into A-fragment layout.
// A-frag (16x32 MxK, wave32): lane m = lane&15; lanes>=16 K += 8;
// V0..3: K=0..7, V4..7: K=16..23 (pairs packed per dword).
// One thread produces one lane's 32 B of one fragment.
// frag id = ((b*16+tt)*3 + oidx)*64 + kc
// =====================================================================
__global__ void __launch_bounds__(256) prep_x_kernel(
    const float* __restrict__ in, unsigned char* __restrict__ XA) {
  unsigned gid = blockIdx.x * 256 + threadIdx.x;
  int lane = gid & 31;
  unsigned frag = gid >> 5;                 // < 98304
  int kc = frag & 63;
  int mt = frag >> 6;                       // (b*16+tt)*3 + oidx
  int bt = mt / 3;
  int o = (mt - bt * 3) - 1;                // shift in {-1,0,1}
  int b = bt >> 4;
  int tcol = ((bt & 15) << 4) + (lane & 15);
  int lhi = (lane & 16) ? 8 : 0;
  unsigned d[8];
#pragma unroll
  for (int j = 0; j < 8; ++j) {
    unsigned half[2];
#pragma unroll
    for (int h = 0; h < 2; ++h) {
      int kl = ((j < 4) ? 2 * j : 8 + 2 * j) + lhi + h;
      int kg = (kc << 5) + kl;              // global feature index d = f*64 + g
      int f = kg >> 6;
      int rc = FLAT.v[kg & 63];
      int rr = ((rc >> 4) - o) & 15;        // roll: rolled[r] = in[(r-o)%16]
      int cc = ((rc & 15) - o) & 15;
      size_t src = ((size_t)(((b * 32 + f) * 16 + rr) * 16 + cc) << 8) + tcol;
      half[h] = f2bf(in[src]);
    }
    d[j] = half[0] | (half[1] << 16);
  }
  uint4* dst = (uint4*)(XA + ((size_t)frag << 10) + lane * 32);
  dst[0] = make_uint4(d[0], d[1], d[2], d[3]);
  dst[1] = make_uint4(d[4], d[5], d[6], d[7]);
}

// =====================================================================
// Kernel 3: fused MLP. One block per (b, 16-t tile): 48 rows (3 offsets x 16 t)
// x 512 hidden. 8 waves, each owns 64 hidden columns (4 N-tiles) x 3 M-tiles.
// GEMM1 (K=2048, bf16 WMMA) -> bias+leaky -> bf16 H1 in LDS ->
// GEMM2 (K=512) -> bias+leaky -> mean over offsets -> LDS transpose -> store.
// =====================================================================
#define H1_STRIDE 1040   // 512*2 + 16 pad (bank spread for ds_load_b128)

__global__ void __launch_bounds__(256) mlp_kernel(
    const unsigned char* __restrict__ XA,
    const unsigned char* __restrict__ WB1,
    const unsigned char* __restrict__ WB2,
    const float* __restrict__ b1v,
    const float* __restrict__ b2v,
    float* __restrict__ out) {
  __shared__ __align__(16) unsigned char sMem[48 * H1_STRIDE];  // 49920 B
  float* sOut = (float*)sMem;   // aliased after barrier: [512][20] fp32 (40960 B)

  const int tid = threadIdx.x;
  const int lane = tid & 31;
  const int w = tid >> 5;
  const int bid = blockIdx.x;           // b*16 + tt
  const int lm = lane & 15;
  const int lhalf = lane >> 4;

  // ---------------- GEMM1 ----------------
  v8f acc[3][4];
#pragma unroll
  for (int o = 0; o < 3; ++o)
#pragma unroll
    for (int nt = 0; nt < 4; ++nt)
      acc[o][nt] = (v8f){0.f, 0.f, 0.f, 0.f, 0.f, 0.f, 0.f, 0.f};

  const unsigned char* aBase = XA + (((size_t)bid * 3) << 16) + lane * 32;
  const unsigned char* bBase1 = WB1 + (((size_t)w * 4) << 16) + lane * 32;
  for (int kc = 0; kc < 64; ++kc) {
    Frag a[3];
#pragma unroll
    for (int o = 0; o < 3; ++o)
      a[o].bf = *(const v16bf*)(aBase + ((size_t)(o * 64 + kc) << 10));
#pragma unroll
    for (int nt = 0; nt < 4; ++nt) {
      Frag bf;
      bf.bf = *(const v16bf*)(bBase1 + ((size_t)(nt * 64 + kc) << 10));
#pragma unroll
      for (int o = 0; o < 3; ++o)
        acc[o][nt] = __builtin_amdgcn_wmma_f32_16x16x32_bf16(
            false, a[o].bf, false, bf.bf, (short)0, acc[o][nt], false, false);
    }
  }

  // bias1 + leaky -> bf16 H1 in LDS: H1[row = o*16+m][k = hidden]
#pragma unroll
  for (int nt = 0; nt < 4; ++nt) {
    int ncol = ((w * 4 + nt) << 4) + lm;
    float bias = b1v[ncol];
#pragma unroll
    for (int o = 0; o < 3; ++o)
#pragma unroll
      for (int v = 0; v < 8; ++v) {
        float h = acc[o][nt][v] + bias;
        h = h > 0.0f ? h : 0.01f * h;
        int row = o * 16 + v + lhalf * 8;
        *(unsigned short*)&sMem[row * H1_STRIDE + ncol * 2] = f2bf(h);
      }
  }
  __syncthreads();

  // ---------------- GEMM2 ----------------
  v8f acc2[3][4];
#pragma unroll
  for (int o = 0; o < 3; ++o)
#pragma unroll
    for (int nt = 0; nt < 4; ++nt)
      acc2[o][nt] = (v8f){0.f, 0.f, 0.f, 0.f, 0.f, 0.f, 0.f, 0.f};

  const unsigned char* bBase2 = WB2 + (((size_t)w * 4) << 14) + lane * 32;
  for (int kc = 0; kc < 16; ++kc) {
    Frag a[3];
#pragma unroll
    for (int o = 0; o < 3; ++o) {
      int row = o * 16 + lm;
      const unsigned char* p = &sMem[row * H1_STRIDE + kc * 64 + lhalf * 16];
      a[o].u[0] = *(const uint4*)p;          // K 0..7 (or 8..15)
      a[o].u[1] = *(const uint4*)(p + 32);   // K 16..23 (or 24..31)
    }
#pragma unroll
    for (int nt = 0; nt < 4; ++nt) {
      Frag bf;
      bf.bf = *(const v16bf*)(bBase2 + ((size_t)(nt * 16 + kc) << 10));
#pragma unroll
      for (int o = 0; o < 3; ++o)
        acc2[o][nt] = __builtin_amdgcn_wmma_f32_16x16x32_bf16(
            false, a[o].bf, false, bf.bf, (short)0, acc2[o][nt], false, false);
    }
  }
  __syncthreads();   // done reading H1; sOut aliases the same LDS

  // bias2 + leaky + mean over offsets -> sOut[n][t] (stride 20 dwords)
#pragma unroll
  for (int nt = 0; nt < 4; ++nt) {
    int ncol = ((w * 4 + nt) << 4) + lm;
    float bias = b2v[ncol];
#pragma unroll
    for (int v = 0; v < 8; ++v) {
      float s = 0.f;
#pragma unroll
      for (int o = 0; o < 3; ++o) {
        float h = acc2[o][nt][v] + bias;
        s += (h > 0.0f ? h : 0.01f * h);
      }
      int t = v + lhalf * 8;
      sOut[ncol * 20 + t] = s * (1.0f / 3.0f);
    }
  }
  __syncthreads();

  // coalesced store: out[b][n][t0..t0+15], each thread handles 2 hidden rows
  int b = bid >> 4, t0 = (bid & 15) << 4;
#pragma unroll
  for (int rep = 0; rep < 2; ++rep) {
    int n = tid + rep * 256;
    const uint4* src = (const uint4*)&sOut[n * 20];
    uint4* dst = (uint4*)(out + (((size_t)b * 512 + n) << 8) + t0);
    dst[0] = src[0]; dst[1] = src[1]; dst[2] = src[2]; dst[3] = src[3];
  }
}

// =====================================================================
extern "C" void kernel_launch(void* const* d_in, const int* in_sizes, int n_in,
                              void* d_out, int out_size, void* d_ws, size_t ws_size,
                              hipStream_t stream) {
  const float* in = (const float*)d_in[0];   // [32,32,16,16,256]
  const float* W1 = (const float*)d_in[1];   // [512,2048]
  const float* b1 = (const float*)d_in[2];   // [512]
  const float* W2 = (const float*)d_in[3];   // [512,512]
  const float* b2 = (const float*)d_in[4];   // [512]
  float* out = (float*)d_out;                // [32,512,256]

  unsigned char* ws = (unsigned char*)d_ws;
  unsigned char* XA = ws;                                  // 98304 frags * 1024 B = 96 MiB
  unsigned char* WB1 = ws + (size_t)98304 * 1024;          // 2 MiB
  unsigned char* WB2 = WB1 + (size_t)32 * 64 * 1024;       // 0.5 MiB

  prep_w_kernel<<<320, 256, 0, stream>>>(W1, W2, WB1, WB2);
  prep_x_kernel<<<12288, 256, 0, stream>>>(in, XA);
  mlp_kernel<<<512, 256, 0, stream>>>(XA, WB1, WB2, b1, b2, out);
}
